// T5Module_79542794322664
// MI455X (gfx1250) — matrix operand order repaired and checked by reading, not compiled, not run on hardware
//
#include <hip/hip_runtime.h>
#include <math.h>
#include <stdint.h>

// ---------------------------------------------------------------------------
// Types for CDNA5 WMMA (wave32): A/B fragments are v16h, C/D fragments v8f.
// ---------------------------------------------------------------------------
typedef _Float16 half8   __attribute__((ext_vector_type(8)));
typedef _Float16 half16  __attribute__((ext_vector_type(16)));
typedef float    floatx8 __attribute__((ext_vector_type(8)));
typedef unsigned int uint4v __attribute__((ext_vector_type(4)));
typedef int      int8v   __attribute__((ext_vector_type(8)));
typedef int      int4v   __attribute__((ext_vector_type(4)));

#define NEGF      (-3.4028234663852886e38f)
#define F16MAXF   65504.0f

__device__ __forceinline__ floatx8 wmma_f16(half16 a, half16 b, floatx8 c) {
  // (neg_a, A, neg_b, B, c_mod, C, reuse_a, reuse_b)
  return __builtin_amdgcn_wmma_f32_16x16x32_f16(false, a, false, b, (short)0, c,
                                                false, false);
}

__device__ __forceinline__ half16 cat8(half8 a, half8 b) {
  half16 r;
#pragma unroll
  for (int i = 0; i < 8; ++i) { r[i] = a[i]; r[i + 8] = b[i]; }
  return r;
}

// ---------------------------------------------------------------------------
// Tensor Data Mover: 2D tile load Global -> LDS (clang-23 6-arg builtin:
// (g0 x4u, g1 x8i, g2 x4i, g3 x4i, x8i, cpol)).
// D# group0: count=1 | lds_addr | global_addr[56:0] | type=2
// D# group1: data_size=2B, pad_enable, pad_interval/amount, tile dims, stride
// LDS row pitch = tile_w + pad (pad inserted by the DMA engine itself).
// ---------------------------------------------------------------------------
__device__ __forceinline__ void tdm_load_2d(const void* gptr, unsigned lds_off,
                                            unsigned tile_w, unsigned tile_h,
                                            unsigned stride_elems,
                                            unsigned pad_interval_code,
                                            unsigned pad_amount_code) {
  const uint64_t ga = (uint64_t)gptr;
  uint4v g0;
  g0[0] = 1u;                                         // count=1 (valid user D#)
  g0[1] = lds_off;                                    // lds_addr (bytes)
  g0[2] = (unsigned)(ga & 0xFFFFFFFFu);               // global_addr[31:0]
  g0[3] = (unsigned)((ga >> 32) & 0x01FFFFFFu) | (2u << 30); // [56:32] | type=2

  int8v g1;
  unsigned d0 = (1u << 16)                            // data_size: 2 bytes
              | (1u << 20)                            // pad_enable
              | (pad_interval_code << 22)             // DWORDs between pads
              | (pad_amount_code << 25);              // pad DWORDs - 1
  g1[0] = (int)d0;
  const unsigned td0 = 0x40000000u, td1 = 0x40000000u; // huge dims: no OOB clamp
  g1[1] = (int)((td0 & 0xFFFFu) << 16);               // tensor_dim0[15:0]
  g1[2] = (int)(((td0 >> 16) & 0xFFFFu) | ((td1 & 0xFFFFu) << 16));
  g1[3] = (int)(((td1 >> 16) & 0xFFFFu) | ((tile_w & 0xFFFFu) << 16)); // tile_dim0
  g1[4] = (int)(tile_h & 0xFFFFu);                    // tile_dim1 (tile_dim2=0)
  g1[5] = (int)stride_elems;                          // tensor_dim0_stride[31:0]
  g1[6] = 0;
  g1[7] = 0;
  const int4v z4 = (int4v)0;                          // groups 2/3 unused (2D)
  const int8v z8 = (int8v)0;
  __builtin_amdgcn_tensor_load_to_lds(g0, g1, z4, z4, z8, 0);
}

__device__ __forceinline__ unsigned lds_offset_of(const void* p) {
  return (unsigned)(uintptr_t)p;                      // low 32 bits = LDS offset
}

// ---------------------------------------------------------------------------
// f32 -> f16 convert (weights, encoder states)
// ---------------------------------------------------------------------------
__global__ void f32_to_f16_kernel(const float* __restrict__ in,
                                  _Float16* __restrict__ out, size_t n) {
  size_t i = (size_t)blockIdx.x * blockDim.x + threadIdx.x;
  if (i < n) out[i] = (_Float16)in[i];
}

// ---------------------------------------------------------------------------
// T5 RMSNorm: one workgroup per row of C=1024 columns; f32 in, f16 out.
// ---------------------------------------------------------------------------
__global__ __launch_bounds__(256)
void rmsnorm_f16_kernel(const float* __restrict__ x, const float* __restrict__ w,
                        _Float16* __restrict__ y, int C) {
  __shared__ float red[8];
  const int row = blockIdx.x;
  const float* xr = x + (size_t)row * C;
  float ss = 0.f;
  for (int c = threadIdx.x; c < C; c += 256) { float v = xr[c]; ss += v * v; }
#pragma unroll
  for (int off = 16; off > 0; off >>= 1) ss += __shfl_xor(ss, off);
  if ((threadIdx.x & 31) == 0) red[threadIdx.x >> 5] = ss;
  __syncthreads();
  float tot = 0.f;
#pragma unroll
  for (int i = 0; i < 8; ++i) tot += red[i];
  const float scale = rsqrtf(tot / (float)C + 1e-6f);
  _Float16* yr = y + (size_t)row * C;
  for (int c = threadIdx.x; c < C; c += 256)
    yr[c] = (_Float16)(w[c] * xr[c] * scale);
}

// ---------------------------------------------------------------------------
// Tiled f16 GEMM with f32 accumulate: C[M,N] = A[M,K] x B[K,N].
// BM=128, BN=128, BK=32. 256 threads = 8 waves; wave grid 4(M) x 2(N).
// A tiles: TDM tensor_load_to_lds, double-buffered (TENSORcnt pipelining).
// B tiles: manual transpose staging in LDS (contiguous B-fragment reads).
// ---------------------------------------------------------------------------
template <bool OUT_F16>
__global__ __launch_bounds__(256)
void gemm_f16_kernel(const _Float16* __restrict__ A, const _Float16* __restrict__ B,
                     void* __restrict__ Cout, int M, int N, int K) {
  constexpr int BM = 128, BN = 128, BK = 32, LDT = BK + 8; // 40 halves = 80B rows
  __shared__ __attribute__((aligned(16))) _Float16 As[2][BM * LDT];
  __shared__ __attribute__((aligned(16))) _Float16 Bt[BN * LDT]; // [col][k]
  const int tid  = threadIdx.x;
  const int wave = tid >> 5;
  const int lane = tid & 31;
  const int l16  = lane & 15;
  const bool hiL = lane >= 16;
  const int m0 = blockIdx.y * BM, n0 = blockIdx.x * BN;
  const int wm = (wave & 3) * 32;   // 2 M-tiles
  const int wn = (wave >> 2) * 64;  // 4 N-tiles

  floatx8 acc[2][4];
#pragma unroll
  for (int i = 0; i < 2; ++i)
#pragma unroll
    for (int j = 0; j < 4; ++j) acc[i][j] = (floatx8)0.0f;

  const _Float16* Arow = A + (size_t)m0 * K;
  const int nT = K / BK;

  // prologue: DMA first A tile (pad_interval=16 DW, pad_amount=4 DW -> LDT=40)
  if (wave == 0)
    tdm_load_2d(Arow, lds_offset_of(&As[0][0]), BK, BM, (unsigned)K, 3, 3);

  for (int it = 0; it < nT; ++it) {
    const int k0 = it * BK;
    // --- stage B tile 32x128, transposed into LDS [col][k] ---
    {
      const int krow = tid >> 3, cc = (tid & 7) * 16;
      const _Float16* src = B + (size_t)(k0 + krow) * N + n0 + cc;
      half8 b0 = *(const half8*)src;
      half8 b1 = *(const half8*)(src + 8);
#pragma unroll
      for (int c = 0; c < 8; ++c) Bt[(cc + c) * LDT + krow] = b0[c];
#pragma unroll
      for (int c = 0; c < 8; ++c) Bt[(cc + 8 + c) * LDT + krow] = b1[c];
      if (k0 + BK < K) __builtin_prefetch(src + (size_t)BK * N, 0, 0);
    }
    // --- DMA next A tile into other buffer; ensure current tile resident ---
    if (wave == 0) {
      if (it + 1 < nT) {
        tdm_load_2d(Arow + (k0 + BK), lds_offset_of(&As[(it + 1) & 1][0]),
                    BK, BM, (unsigned)K, 3, 3);
        __builtin_amdgcn_s_wait_tensorcnt(1);  // oldest (current tile) done
      } else {
        __builtin_amdgcn_s_wait_tensorcnt(0);
      }
    }
    __syncthreads();

    const _Float16* Asb = &As[it & 1][0];
    // --- fragments (ISA 16-bit A/B VGPR layouts) ---
    half16 afrag[2], bfrag[4];
    const int ka = hiL ? 8 : 0; // lo lanes K{0..7,16..23}, hi lanes K{8..15,24..31}
#pragma unroll
    for (int mt = 0; mt < 2; ++mt) {
      const _Float16* ap = &Asb[(wm + mt * 16 + l16) * LDT];
      afrag[mt] = cat8(*(const half8*)(ap + ka), *(const half8*)(ap + 16 + ka));
    }
    const int kb = hiL ? 8 : 0; // lo lanes K{0..7,16..23}, hi lanes K{8..15,24..31}
#pragma unroll
    for (int nt = 0; nt < 4; ++nt) {
      const _Float16* bp = &Bt[(wn + nt * 16 + l16) * LDT] + kb;
      bfrag[nt] = cat8(*(const half8*)bp, *(const half8*)(bp + 16));
    }
#pragma unroll
    for (int mt = 0; mt < 2; ++mt)
#pragma unroll
      for (int nt = 0; nt < 4; ++nt)
        acc[mt][nt] = wmma_f16(afrag[mt], bfrag[nt], acc[mt][nt]);
    __syncthreads(); // all reads done before next B staging / A DMA overwrite
  }

  // --- epilogue: C layout VGPR r, lanes 0-15 -> M=r, lanes 16-31 -> M=r+8 ---
#pragma unroll
  for (int mt = 0; mt < 2; ++mt)
#pragma unroll
    for (int nt = 0; nt < 4; ++nt) {
      const int col = n0 + wn + nt * 16 + l16;
#pragma unroll
      for (int r = 0; r < 8; ++r) {
        const int row = m0 + wm + mt * 16 + r + (hiL ? 8 : 0);
        if (OUT_F16)
          ((_Float16*)Cout)[(size_t)row * N + col] = (_Float16)acc[mt][nt][r];
        else
          ((float*)Cout)[(size_t)row * N + col] = acc[mt][nt][r];
      }
    }
}

// ---------------------------------------------------------------------------
// T5 relative-position bias (unidirectional buckets), rel_emb[32][16] f32.
// ---------------------------------------------------------------------------
__device__ __forceinline__ float rel_bias(const float* __restrict__ rel_emb,
                                          int h, int i, int j) {
  int rp = i - j;                 // -min(j-i, 0)
  if (rp < 0) rp = 0;
  int bucket;
  if (rp < 16) {
    bucket = rp;
  } else {
    float f = __logf((float)rp * (1.0f / 16.0f)) * (16.0f / 2.0794415416798357f);
    bucket = 16 + (int)f;
    if (bucket > 31) bucket = 31;
  }
  return rel_emb[bucket * 16 + h];
}

// ---------------------------------------------------------------------------
// Flash attention, T5 style (no 1/sqrt(d)). q,k,v,out: f16 [B, L, 16, 64].
// Block = 128 threads (4 waves); each wave owns a 16-query tile; grid covers
// B * heads * (L/64). Online softmax over 32-key blocks. V blocks arrive by
// TDM DMA, double-buffered on TENSORcnt.
// ---------------------------------------------------------------------------
__global__ __launch_bounds__(128)
void attention_kernel(const _Float16* __restrict__ q, const _Float16* __restrict__ k,
                      const _Float16* __restrict__ v, const int* __restrict__ mask,
                      const float* __restrict__ rel_emb, _Float16* __restrict__ out,
                      int B, int L, int useBias) {
  constexpr int NH = 16, HD = 64, AW = NH * HD;      // AW = 1024
  constexpr int LDV = HD + 8;                        // 72 halves (pad by DMA)
  __shared__ __attribute__((aligned(16))) _Float16 Vs[2][32 * LDV];
  __shared__ __attribute__((aligned(16))) _Float16 Ps[4][16 * 32];

  const int qtiles = L / 64;
  const int blk = blockIdx.x;
  const int b   = blk / (NH * qtiles);
  const int rem = blk % (NH * qtiles);
  const int h   = rem / qtiles;
  const int qb  = rem % qtiles;
  const int wave = threadIdx.x >> 5;
  const int lane = threadIdx.x & 31;
  const int l16  = lane & 15;
  const bool hiL = lane >= 16;
  const int q0 = qb * 64 + wave * 16;
  const size_t base = (size_t)b * L * AW + (size_t)h * HD;

  // Q fragments for the two 32-wide K chunks of head dim 64
  half16 qf[2];
#pragma unroll
  for (int c = 0; c < 2; ++c) {
    const _Float16* qp = q + base + (size_t)(q0 + l16) * AW + c * 32 + (hiL ? 8 : 0);
    qf[c] = cat8(*(const half8*)qp, *(const half8*)(qp + 16));
  }

  float mrow[8], lrow[8];
  floatx8 o[4];
#pragma unroll
  for (int r = 0; r < 8; ++r) { mrow[r] = -INFINITY; lrow[r] = 0.f; }
#pragma unroll
  for (int t = 0; t < 4; ++t) o[t] = (floatx8)0.0f;

  const int nKB = L / 32;
  // prologue: DMA first V block [32 x 64] (pad_interval=32 DW, amount=4 DW)
  if (wave == 0)
    tdm_load_2d(v + base, lds_offset_of(&Vs[0][0]), HD, 32, AW, 4, 3);

  for (int it = 0; it < nKB; ++it) {
    const int key0 = it * 32;
    if (wave == 0) {
      if (it + 1 < nKB) {
        tdm_load_2d(v + base + (size_t)(key0 + 32) * AW,
                    lds_offset_of(&Vs[(it + 1) & 1][0]), HD, 32, AW, 4, 3);
        __builtin_amdgcn_s_wait_tensorcnt(1);
      } else {
        __builtin_amdgcn_s_wait_tensorcnt(0);
      }
    }
    __syncthreads();
    const _Float16* Vb = &Vs[it & 1][0];

    // ---- S = Q @ K^T (two 16x16 fragments over 32 keys) ----
    floatx8 s[2];
#pragma unroll
    for (int g = 0; g < 2; ++g) {
      s[g] = (floatx8)0.0f;
#pragma unroll
      for (int c = 0; c < 2; ++c) {
        const int key = key0 + g * 16 + l16;
        // B-layout: lane = key column; halves = d{0..7,16..23} (+8 for hi lanes)
        const _Float16* kp = k + base + (size_t)key * AW + c * 32 + (hiL ? 8 : 0);
        half16 kf = cat8(*(const half8*)kp, *(const half8*)(kp + 16));
        s[g] = wmma_f16(qf[c], kf, s[g]);
      }
    }

    // ---- bias + mask ----
#pragma unroll
    for (int g = 0; g < 2; ++g) {
      const int j = key0 + g * 16 + l16;
      const int mk = mask[b * L + j];
#pragma unroll
      for (int r = 0; r < 8; ++r) {
        const int i = q0 + r + (hiL ? 8 : 0);
        float val = s[g][r];
        if (useBias) val += rel_bias(rel_emb, h, i, j);
        s[g][r] = (mk == 0) ? NEGF : val;
      }
    }

    // ---- online softmax (row stats via 16-lane shuffle reduce) ----
    float scl[8];
#pragma unroll
    for (int r = 0; r < 8; ++r) {
      float bm = fmaxf(s[0][r], s[1][r]);
#pragma unroll
      for (int off = 8; off > 0; off >>= 1) bm = fmaxf(bm, __shfl_xor(bm, off));
      const float mn = fmaxf(mrow[r], bm);
      const float sc = __expf(mrow[r] - mn);
      const float p0 = __expf(s[0][r] - mn);
      const float p1 = __expf(s[1][r] - mn);
      float rs = p0 + p1;
#pragma unroll
      for (int off = 8; off > 0; off >>= 1) rs += __shfl_xor(rs, off);
      lrow[r] = lrow[r] * sc + rs;
      mrow[r] = mn;
      scl[r]  = sc;
      s[0][r] = p0;
      s[1][r] = p1;
    }
#pragma unroll
    for (int t = 0; t < 4; ++t)
#pragma unroll
      for (int r = 0; r < 8; ++r) o[t][r] *= scl[r];

    // ---- reshape P (C-layout) -> A-layout via per-wave LDS tile ----
    _Float16* pp = &Ps[wave][0];
#pragma unroll
    for (int r = 0; r < 8; ++r) {
      const int row = r + (hiL ? 8 : 0);
      pp[row * 32 + l16]      = (_Float16)s[0][r];
      pp[row * 32 + 16 + l16] = (_Float16)s[1][r];
    }
    asm volatile("s_wait_dscnt 0" ::: "memory"); // same-wave DS RAW fence

    half16 pf;
    {
      const _Float16* pr = &Ps[wave][l16 * 32] + (hiL ? 8 : 0);
      pf = cat8(*(const half8*)pr, *(const half8*)(pr + 16));
    }

    // ---- O += P @ V ----
    const int kkb = hiL ? 8 : 0;
#pragma unroll
    for (int t = 0; t < 4; ++t) {
      half16 vf;
#pragma unroll
      for (int kk = 0; kk < 16; ++kk)
        vf[kk] = Vb[(kkb + kk + (kk >= 8 ? 8 : 0)) * LDV + t * 16 + l16];
      o[t] = wmma_f16(pf, vf, o[t]);
    }
    __syncthreads(); // reads done before DMA overwrites this buffer
  }

  // ---- normalize and store ----
#pragma unroll
  for (int t = 0; t < 4; ++t)
#pragma unroll
    for (int r = 0; r < 8; ++r) {
      const int row = q0 + r + (hiL ? 8 : 0);
      const float denom = (lrow[r] > 0.f) ? lrow[r] : 1.f;
      out[base + (size_t)row * AW + t * 16 + l16] = (_Float16)(o[t][r] / denom);
    }
}

// ---------------------------------------------------------------------------
// Elementwise: x = clamp(x + t, +/- 65504)
// ---------------------------------------------------------------------------
__global__ void add_clamp_kernel(float* __restrict__ x, const float* __restrict__ t,
                                 size_t n) {
  size_t i = (size_t)blockIdx.x * blockDim.x + threadIdx.x;
  if (i < n) {
    float v = x[i] + t[i];
    x[i] = fminf(fmaxf(v, -F16MAXF), F16MAXF);
  }
}

// ---------------------------------------------------------------------------
// GEGLU: g1 = g1 * gelu_tanh(g2), f16 in/out, f32 math.
// ---------------------------------------------------------------------------
__global__ void geglu_kernel(_Float16* __restrict__ g1, const _Float16* __restrict__ g2,
                             size_t n) {
  size_t i = (size_t)blockIdx.x * blockDim.x + threadIdx.x;
  if (i < n) {
    float a = (float)g1[i];
    float x = (float)g2[i];
    float g = 0.5f * x * (1.0f + tanhf(0.7978845608028654f * (x + 0.044715f * x * x * x)));
    g1[i] = (_Float16)(a * g);
  }
}

// ---------------------------------------------------------------------------
// Orchestration
// ---------------------------------------------------------------------------
extern "C" void kernel_launch(void* const* d_in, const int* in_sizes, int n_in,
                              void* d_out, int out_size, void* d_ws, size_t ws_size,
                              hipStream_t stream) {
  (void)in_sizes; (void)n_in; (void)out_size; (void)ws_size;
  const int B = 4, L = 1024, C = 1024, A = 1024, F = 4096;
  const int M = B * L; // 4096 rows
  const int NH = 16;

  const float* x_in   = (const float*)d_in[0];
  const float* enc    = (const float*)d_in[1];
  const float* n1w    = (const float*)d_in[2];
  const float* n2w    = (const float*)d_in[3];
  const float* n3w    = (const float*)d_in[4];
  const float* w_f32[11] = {
      (const float*)d_in[5],  (const float*)d_in[6],  (const float*)d_in[7],
      (const float*)d_in[8],  (const float*)d_in[9],  (const float*)d_in[10],
      (const float*)d_in[11], (const float*)d_in[12], (const float*)d_in[13],
      (const float*)d_in[14], (const float*)d_in[15]};
  const size_t w_elems[11] = {
      (size_t)C * A, (size_t)C * A, (size_t)C * A, (size_t)A * C,
      (size_t)C * A, (size_t)C * A, (size_t)C * A, (size_t)A * C,
      (size_t)C * F, (size_t)C * F, (size_t)F * C};
  const float* rel_emb = (const float*)d_in[16];
  const int* mask      = (const int*)d_in[17];
  const int* enc_mask  = (const int*)d_in[18];
  float* out = (float*)d_out;

  // ---- workspace carve-out (256B aligned) ----
  char* ws = (char*)d_ws;
  size_t off = 0;
  auto carve = [&](size_t bytes) -> void* {
    void* p = ws + off;
    off += (bytes + 255) & ~(size_t)255;
    return p;
  };
  _Float16* w16[11];
  for (int i = 0; i < 11; ++i) w16[i] = (_Float16*)carve(w_elems[i] * 2);
  _Float16* enc16 = (_Float16*)carve((size_t)M * C * 2);
  _Float16* xn16  = (_Float16*)carve((size_t)M * C * 2);
  _Float16* q16   = (_Float16*)carve((size_t)M * A * 2);
  _Float16* k16   = (_Float16*)carve((size_t)M * A * 2);
  _Float16* v16   = (_Float16*)carve((size_t)M * A * 2);
  _Float16* ctx16 = (_Float16*)carve((size_t)M * A * 2);
  _Float16* g1    = (_Float16*)carve((size_t)M * F * 2);
  _Float16* g2    = (_Float16*)carve((size_t)M * F * 2);
  float*    tmp   = (float*)carve((size_t)M * C * 4);

  auto cvt = [&](const float* src, _Float16* dst, size_t n) {
    f32_to_f16_kernel<<<(unsigned)((n + 255) / 256), 256, 0, stream>>>(src, dst, n);
  };
  auto gemm16 = [&](const _Float16* Ax, const _Float16* Bx, _Float16* Cx,
                    int m, int n, int kk) {
    dim3 g(n / 128, m / 128);
    gemm_f16_kernel<true><<<g, 256, 0, stream>>>(Ax, Bx, (void*)Cx, m, n, kk);
  };
  auto gemm32 = [&](const _Float16* Ax, const _Float16* Bx, float* Cx,
                    int m, int n, int kk) {
    dim3 g(n / 128, m / 128);
    gemm_f16_kernel<false><<<g, 256, 0, stream>>>(Ax, Bx, (void*)Cx, m, n, kk);
  };

  // running residual lives in d_out (f32)
  (void)hipMemcpyAsync(out, x_in, (size_t)M * C * 4, hipMemcpyDeviceToDevice,
                       stream);

  // weight / encoder converts
  for (int i = 0; i < 11; ++i) cvt(w_f32[i], w16[i], w_elems[i]);
  cvt(enc, enc16, (size_t)M * C);

  const size_t nMC = (size_t)M * C;
  const unsigned gMC = (unsigned)((nMC + 255) / 256);
  const int attnBlocks = B * NH * (L / 64);

  // ---- self-attention ----
  rmsnorm_f16_kernel<<<M, 256, 0, stream>>>(out, n1w, xn16, C);
  gemm16(xn16, w16[0], q16, M, A, C);
  gemm16(xn16, w16[1], k16, M, A, C);
  gemm16(xn16, w16[2], v16, M, A, C);
  attention_kernel<<<attnBlocks, 128, 0, stream>>>(q16, k16, v16, mask, rel_emb,
                                                   ctx16, B, L, /*useBias=*/1);
  gemm32(ctx16, w16[3], tmp, M, C, A);
  add_clamp_kernel<<<gMC, 256, 0, stream>>>(out, tmp, nMC);

  // ---- cross-attention (K/V from raw encoder states) ----
  rmsnorm_f16_kernel<<<M, 256, 0, stream>>>(out, n2w, xn16, C);
  gemm16(xn16, w16[4], q16, M, A, C);
  gemm16(enc16, w16[5], k16, M, A, C);
  gemm16(enc16, w16[6], v16, M, A, C);
  attention_kernel<<<attnBlocks, 128, 0, stream>>>(q16, k16, v16, enc_mask, rel_emb,
                                                   ctx16, B, L, /*useBias=*/0);
  gemm32(ctx16, w16[7], tmp, M, C, A);
  add_clamp_kernel<<<gMC, 256, 0, stream>>>(out, tmp, nMC);

  // ---- GEGLU FFN ----
  rmsnorm_f16_kernel<<<M, 256, 0, stream>>>(out, n3w, xn16, C);
  gemm16(xn16, w16[8], g1, M, F, C);   // h @ fc1
  gemm16(xn16, w16[9], g2, M, F, C);   // h @ gate
  {
    const size_t nMF = (size_t)M * F;
    geglu_kernel<<<(unsigned)((nMF + 255) / 256), 256, 0, stream>>>(g1, g2, nMF);
  }
  gemm32(g1, w16[10], tmp, M, C, F);   // @ fc2
  add_clamp_kernel<<<gMC, 256, 0, stream>>>(out, tmp, nMC);
}
